// BotNet50_79852031967978
// MI455X (gfx1250) — compile-verified, matrix operand order checked by reading, not run on hardware
//
#include <hip/hip_runtime.h>

// ---------------------------------------------------------------------------
// BotNet relative-position self-attention for MI455X (gfx1250, wave32, WMMA)
// B=16, C=512, H=W=32, L=1024, HEADS=4, DQK=DV=128
//
// All GEMM stages use v_wmma_f32_16x16x32_bf16 (bf16 inputs, f32 accum).
// logits[(qh,qw),(kh,kw)] = q.k + q.rel_w[kw-qw+31] + q.rel_h[kh-qh+31]
// RW/RH are precomputed per query position (L2-resident), attention is
// flash-style streaming (no 268MB logits tensor ever materialized).
// f32->bf16 uses the native converter (v_cvt_pk_bf16_f32 / v_cvt_bf16_f32).
// ---------------------------------------------------------------------------

typedef __attribute__((ext_vector_type(16))) __bf16 v16bf;
typedef __attribute__((ext_vector_type(2)))  __bf16 v2bf;
typedef __attribute__((ext_vector_type(8)))  float  v8f;

struct U4 { unsigned int a, b, c, d; };   // POD 128-bit (safe inside union)
struct F4 { float x, y, z, w; };

union AFrag {
    v16bf        bf;
    unsigned int u[8];
    U4           q[2];
};

// Native f32->bf16 (RNE on gfx1250 hardware converters).
__device__ __forceinline__ unsigned short f2bf(float f) {
    return __builtin_bit_cast(unsigned short, (__bf16)f);
}
__device__ __forceinline__ unsigned int pk(float a, float b) {
#if __has_builtin(__builtin_amdgcn_cvt_pk_bf16_f32)
    v2bf p = __builtin_amdgcn_cvt_pk_bf16_f32(a, b);
    return __builtin_bit_cast(unsigned int, p);
#else
    v2bf p;
    p[0] = (__bf16)a;
    p[1] = (__bf16)b;
    return __builtin_bit_cast(unsigned int, p);
#endif
}

// A/B fragment from bf16 row-major memory. Lane m (m&15 = matrix row/col),
// kbase = 8*(lane>=16); VGPR0-3: K = kbase+0..7, VGPR4-7: K = kbase+16..23.
__device__ __forceinline__ AFrag ld_bf16_rows(const unsigned short* __restrict__ base,
                                              int row, int stride, int k0, int lane) {
    AFrag f;
    int half = lane >> 4;
    const U4* p = (const U4*)(base + (size_t)row * stride + k0);
    f.q[0] = p[half];
    f.q[1] = p[half + 2];
    return f;
}

// Fragment from f32 row-major memory (convert to bf16 in-register).
__device__ __forceinline__ AFrag ld_f32_rows(const float* __restrict__ base,
                                             int row, int stride, int k0, int lane) {
    AFrag f;
    int half = lane >> 4;
    const float* r = base + (size_t)row * stride + k0 + 8 * half;
    F4 a0 = ((const F4*)r)[0];
    F4 a1 = ((const F4*)r)[1];
    F4 b0 = ((const F4*)(r + 16))[0];
    F4 b1 = ((const F4*)(r + 16))[1];
    f.u[0] = pk(a0.x, a0.y); f.u[1] = pk(a0.z, a0.w);
    f.u[2] = pk(a1.x, a1.y); f.u[3] = pk(a1.z, a1.w);
    f.u[4] = pk(b0.x, b0.y); f.u[5] = pk(b0.z, b0.w);
    f.u[6] = pk(b1.x, b1.y); f.u[7] = pk(b1.z, b1.w);
    return f;
}

// Fragment from f32 column-strided memory: element K at base[(k0+K)*rowstride + col].
__device__ __forceinline__ AFrag ld_f32_cols(const float* __restrict__ base,
                                             int col, int rowstride, int k0, int lane) {
    AFrag f;
    int kb = k0 + 8 * (lane >> 4);
#pragma unroll
    for (int i = 0; i < 4; ++i) {
        int c = kb + 2 * i;
        f.u[i]     = pk(base[(size_t)c * rowstride + col],
                        base[(size_t)(c + 1) * rowstride + col]);
        f.u[i + 4] = pk(base[(size_t)(c + 16) * rowstride + col],
                        base[(size_t)(c + 17) * rowstride + col]);
    }
    return f;
}

__device__ __forceinline__ v8f wmma_bf16(const AFrag& A, const AFrag& B, v8f C) {
    return __builtin_amdgcn_wmma_f32_16x16x32_bf16(
        /*neg_a=*/false, A.bf, /*neg_b=*/false, B.bf,
        /*c_mod=*/(short)0, C, /*reuse_a=*/false, /*reuse_b=*/false);
}

// ---------------------------------------------------------------------------
// Kernel 1: 1x1-conv projections as GEMM.  C[M,L] = W[M,512] * X[512,L] per batch.
// mode 0: W = w_qk (M=1024). rows<512 -> Q (scaled, bf16 [bh][l][d]),
//                            rows>=512 -> K (bf16 [bh][l][d]).
// mode 1: W = w_v  (M=512)  -> V transposed (bf16 [bh][d][l]).
// ---------------------------------------------------------------------------
__global__ __launch_bounds__(128) void k_proj(const float* __restrict__ X,
                                              const float* __restrict__ W,
                                              unsigned short* __restrict__ outQ,
                                              unsigned short* __restrict__ outK,
                                              unsigned short* __restrict__ outVt,
                                              int mode, float qscale) {
    int lane  = threadIdx.x & 31;
    int wave  = threadIdx.x >> 5;
    int b     = blockIdx.z;
    int obase = blockIdx.x * 16;
    int lbase = blockIdx.y * 64 + wave * 16;
    const float* Xb = X + (size_t)b * 512 * 1024;

    v8f acc = {};
    for (int c0 = 0; c0 < 512; c0 += 32) {
        AFrag a  = ld_f32_rows(W,  obase + (lane & 15), 512, c0, lane);
        AFrag bf = ld_f32_cols(Xb, lbase + (lane & 15), 1024, c0, lane);
        acc = wmma_bf16(a, bf, acc);
    }

    int col = lane & 15;
    int rhi = (lane >> 4) * 8;
    int o0  = obase + rhi;           // 8 consecutive output rows per lane
    int l   = lbase + col;

    if (mode == 0) {
        bool isQ = (o0 < 512);
        int  oo  = isQ ? o0 : o0 - 512;
        int  head = oo >> 7, d0 = oo & 127;
        float s = isQ ? qscale : 1.0f;
        unsigned short* dst =
            (isQ ? outQ : outK) + (((size_t)(b * 4 + head) * 1024 + l) * 128 + d0);
        U4 r;
        r.a = pk(acc[0] * s, acc[1] * s);
        r.b = pk(acc[2] * s, acc[3] * s);
        r.c = pk(acc[4] * s, acc[5] * s);
        r.d = pk(acc[6] * s, acc[7] * s);
        *(U4*)dst = r;                               // 8 bf16, 16B aligned
    } else {
        int head = o0 >> 7, d0 = o0 & 127;
#pragma unroll
        for (int v = 0; v < 8; ++v)
            outVt[((size_t)(b * 4 + head) * 128 + d0 + v) * 1024 + l] = f2bf(acc[v]);
    }
}

// ---------------------------------------------------------------------------
// Kernel 2: relative-logit tables.
// RW[bh][q][m] = q_vec . rel_width[m],  RH[bh][q][m] = q_vec . rel_height[m]
// (m padded 63 -> 64; column 63 is never read by k_attn).
// ---------------------------------------------------------------------------
__global__ __launch_bounds__(128) void k_rel(const unsigned short* __restrict__ Qbf,
                                             const float* __restrict__ relW,
                                             const float* __restrict__ relH,
                                             float* __restrict__ RW,
                                             float* __restrict__ RH) {
    int lane  = threadIdx.x & 31;
    int wave  = threadIdx.x >> 5;
    int bh    = blockIdx.y;
    int lbase = blockIdx.x * 64 + wave * 16;
    const unsigned short* Qb = Qbf + (size_t)bh * 1024 * 128;

    AFrag qf[4];
#pragma unroll
    for (int i = 0; i < 4; ++i)
        qf[i] = ld_bf16_rows(Qb, lbase + (lane & 15), 128, i * 32, lane);

    int col = lane & 15;
    int rhi = (lane >> 4) * 8;

#pragma unroll
    for (int dir = 0; dir < 2; ++dir) {
        const float* rel = dir ? relH : relW;
        float*       out = dir ? RH : RW;
        for (int mt = 0; mt < 4; ++mt) {
            int mrow = mt * 16 + col;
            if (mrow > 62) mrow = 62;                // clamp pad row (unread)
            v8f acc = {};
#pragma unroll
            for (int kc = 0; kc < 4; ++kc) {
                AFrag rb = ld_f32_rows(rel, mrow, 128, kc * 32, lane);
                acc = wmma_bf16(qf[kc], rb, acc);
            }
            int m = mt * 16 + col;
#pragma unroll
            for (int v = 0; v < 8; ++v)
                out[((size_t)bh * 1024 + lbase + rhi + v) * 64 + m] = acc[v];
        }
    }
}

// ---------------------------------------------------------------------------
// Kernel 3: flash attention. 4 waves/block, 16 query rows per wave,
// stream 32 keys per iteration. S and O via WMMA, softmax online.
// ---------------------------------------------------------------------------
__global__ __launch_bounds__(128) void k_attn(const unsigned short* __restrict__ Qbf,
                                              const unsigned short* __restrict__ Kbf,
                                              const unsigned short* __restrict__ Vtb,
                                              const float* __restrict__ RWg,
                                              const float* __restrict__ RHg,
                                              float* __restrict__ out) {
    __shared__ alignas(16) unsigned short pb[4][16][32];   // per-wave P bounce
    int lane  = threadIdx.x & 31;
    int wave  = threadIdx.x >> 5;
    int bh    = blockIdx.y;
    int qbase = blockIdx.x * 64 + wave * 16;

    const unsigned short* Qb = Qbf + (size_t)bh * 1024 * 128;
    const unsigned short* Kb = Kbf + (size_t)bh * 1024 * 128;
    const unsigned short* Vb = Vtb + (size_t)bh * 128 * 1024;
    const float* RW = RWg + (size_t)bh * 1024 * 64;
    const float* RH = RHg + (size_t)bh * 1024 * 64;

    AFrag qf[4];
#pragma unroll
    for (int i = 0; i < 4; ++i)
        qf[i] = ld_bf16_rows(Qb, qbase + (lane & 15), 128, i * 32, lane);

    v8f zero = {};
    v8f o[8];
#pragma unroll
    for (int i = 0; i < 8; ++i) o[i] = zero;
    float mrow[8], lrow[8];
#pragma unroll
    for (int v = 0; v < 8; ++v) { mrow[v] = -1e30f; lrow[v] = 0.0f; }

    int col = lane & 15;
    int rhi = (lane >> 4) * 8;

    for (int kb = 0; kb < 1024; kb += 32) {
        if (kb + 32 < 1024) __builtin_prefetch(Kb + (size_t)(kb + 32) * 128, 0, 0);

        // S = Q . K^T for keys [kb, kb+32): two 16x16 f32 tiles
        v8f s0 = zero, s1 = zero;
#pragma unroll
        for (int kc = 0; kc < 4; ++kc) {
            AFrag kf0 = ld_bf16_rows(Kb, kb + col,      128, kc * 32, lane);
            s0 = wmma_bf16(qf[kc], kf0, s0);
            AFrag kf1 = ld_bf16_rows(Kb, kb + 16 + col, 128, kc * 32, lane);
            s1 = wmma_bf16(qf[kc], kf1, s1);
        }

        // add relative-position logits (L2-resident gathers)
#pragma unroll
        for (int v = 0; v < 8; ++v) {
            int q  = qbase + rhi + v;
            int qh = q >> 5, qw = q & 31;
            int k0 = kb + col, k1 = kb + 16 + col;
            s0[v] += RW[(size_t)q * 64 + ((k0 & 31) - qw + 31)]
                   + RH[(size_t)q * 64 + ((k0 >> 5) - qh + 31)];
            s1[v] += RW[(size_t)q * 64 + ((k1 & 31) - qw + 31)]
                   + RH[(size_t)q * 64 + ((k1 >> 5) - qh + 31)];
        }

        // online softmax: 16-lane row reductions (rows live across a lane half)
        float alp[8];
#pragma unroll
        for (int v = 0; v < 8; ++v) {
            float t = fmaxf(s0[v], s1[v]);
#pragma unroll
            for (int mk = 1; mk < 16; mk <<= 1) t = fmaxf(t, __shfl_xor(t, mk, 16));
            float mn = fmaxf(mrow[v], t);
            alp[v]   = __expf(mrow[v] - mn);
            mrow[v]  = mn;
            float e0 = __expf(s0[v] - mn);
            float e1 = __expf(s1[v] - mn);
            s0[v] = e0; s1[v] = e1;
            float rs = e0 + e1;
#pragma unroll
            for (int mk = 1; mk < 16; mk <<= 1) rs += __shfl_xor(rs, mk, 16);
            lrow[v] = lrow[v] * alp[v] + rs;
        }
#pragma unroll
        for (int t = 0; t < 8; ++t)
#pragma unroll
            for (int v = 0; v < 8; ++v) o[t][v] *= alp[v];

        // C-layout -> A-layout for P via per-wave LDS bounce (in-order DS)
#pragma unroll
        for (int v = 0; v < 8; ++v) {
            pb[wave][rhi + v][col]      = f2bf(s0[v]);
            pb[wave][rhi + v][16 + col] = f2bf(s1[v]);
        }
        asm volatile("s_wait_dscnt 0" ::: "memory");
        AFrag pf;
        {
            int half = lane >> 4;
            const U4* pp = (const U4*)&pb[wave][lane & 15][0];
            pf.q[0] = pp[half];
            pf.q[1] = pp[half + 2];
        }

        // O += P . V  (V stored transposed: contiguous per-lane B fragments)
#pragma unroll
        for (int t = 0; t < 8; ++t) {
            AFrag vf = ld_bf16_rows(Vb, t * 16 + col, 1024, kb, lane);
            o[t] = wmma_bf16(pf, vf, o[t]);
        }
    }

    // epilogue: divide by row sum, write [B, head*128+dv, H, W]
    int b = bh >> 2, head = bh & 3;
#pragma unroll
    for (int t = 0; t < 8; ++t) {
        int dv = t * 16 + col;
#pragma unroll
        for (int v = 0; v < 8; ++v) {
            int q = qbase + rhi + v;
            out[((size_t)(b * 512 + head * 128 + dv)) * 1024 + q] = o[t][v] / lrow[v];
        }
    }
}

// ---------------------------------------------------------------------------
extern "C" void kernel_launch(void* const* d_in, const int* in_sizes, int n_in,
                              void* d_out, int out_size, void* d_ws, size_t ws_size,
                              hipStream_t stream) {
    (void)in_sizes; (void)n_in; (void)out_size; (void)ws_size;
    const float* fm    = (const float*)d_in[0];   // [16,512,32,32]
    const float* w_qk  = (const float*)d_in[1];   // [1024,512]
    const float* w_v   = (const float*)d_in[2];   // [512,512]
    const float* rel_h = (const float*)d_in[3];   // [63,128]
    const float* rel_w = (const float*)d_in[4];   // [63,128]
    float* out = (float*)d_out;

    char* ws = (char*)d_ws;
    const size_t szQKV = (size_t)16 * 4 * 1024 * 128 * 2;   // 16 MiB bf16 each
    const size_t szREL = (size_t)16 * 4 * 1024 * 64 * 4;    // 16 MiB f32 each
    unsigned short* Qbf = (unsigned short*)(ws);
    unsigned short* Kbf = (unsigned short*)(ws + szQKV);
    unsigned short* Vt  = (unsigned short*)(ws + 2 * szQKV);
    float* RW = (float*)(ws + 3 * szQKV);
    float* RH = (float*)(ws + 3 * szQKV + szREL);

    const float qscale = 0.08838834764831845f;   // 128^-0.5

    k_proj<<<dim3(64, 16, 16), 128, 0, stream>>>(fm, w_qk, Qbf, Kbf, nullptr, 0, qscale);
    k_proj<<<dim3(32, 16, 16), 128, 0, stream>>>(fm, w_v, nullptr, nullptr, Vt, 1, 1.0f);
    k_rel <<<dim3(16, 64),     128, 0, stream>>>(Qbf, rel_w, rel_h, RW, RH);
    k_attn<<<dim3(16, 64),     128, 0, stream>>>(Qbf, Kbf, Vt, RW, RH, out);
}
